// InfoNCE_Loss_27075473834562
// MI455X (gfx1250) — matrix operand order repaired and checked by reading, not compiled, not run on hardware
//
#include <hip/hip_runtime.h>
#include <hip/hip_bf16.h>

// ---------------------------------------------------------------------------
// Problem constants (fixed by setup_inputs)
// ---------------------------------------------------------------------------
#define B_DIM   32
#define C_DIM   512
#define G_DIM   16
#define P_STEPS 5
#define NEG     16
#define SKIP    1
#define EPS_F   1e-11f

typedef __attribute__((ext_vector_type(16))) __bf16 v16bf;
typedef __attribute__((ext_vector_type(8)))  float  v8f;

// Stage-k row count: Gp = G - (k + SKIP) = 15 - k ; Nk = Gp * G * B = (15-k)*512
__device__ __host__ inline int stage_rows(int kk) { return (G_DIM - (kk + SKIP)) * G_DIM * B_DIM; }

// ---------------------------------------------------------------------------
// threefry2x32 (JAX-style key = (0, 1000+k), counter = (0, t))
// ---------------------------------------------------------------------------
__device__ inline unsigned int rotl32(unsigned int x, int n) {
    return (x << n) | (x >> (32 - n));
}
__device__ inline unsigned int threefry2x32(unsigned int k0, unsigned int k1,
                                            unsigned int c0, unsigned int c1) {
    const unsigned int ks2 = k0 ^ k1 ^ 0x1BD11BDAu;
    unsigned int x0 = c0 + k0, x1 = c1 + k1;
    const int rA[4] = {13, 15, 26, 6};
    const int rB[4] = {17, 29, 16, 24};
#pragma unroll
    for (int r = 0; r < 4; ++r) { x0 += x1; x1 = rotl32(x1, rA[r]); x1 ^= x0; }
    x0 += k1; x1 += ks2 + 1u;
#pragma unroll
    for (int r = 0; r < 4; ++r) { x0 += x1; x1 = rotl32(x1, rB[r]); x1 ^= x0; }
    x0 += ks2; x1 += k0 + 2u;
#pragma unroll
    for (int r = 0; r < 4; ++r) { x0 += x1; x1 = rotl32(x1, rA[r]); x1 ^= x0; }
    x0 += k0; x1 += k1 + 3u;
#pragma unroll
    for (int r = 0; r < 4; ++r) { x0 += x1; x1 = rotl32(x1, rB[r]); x1 ^= x0; }
    x0 += k1; x1 += ks2 + 4u;
    return x0 ^ x1;
}

// ---------------------------------------------------------------------------
// d_out zero-init (harness poisons it with 0xAA)
// ---------------------------------------------------------------------------
__global__ void init_out(float* out) { out[0] = 0.0f; }

// ---------------------------------------------------------------------------
// Transpose (B,C,G,G) -> (B,G,G,C): z to bf16, c to f32. Coalesced writes.
// grid = (C/256, G*G, B), block = 256
// ---------------------------------------------------------------------------
__global__ __launch_bounds__(256) void convert_zc(const float* __restrict__ z,
                                                  const float* __restrict__ c,
                                                  __bf16* __restrict__ zt,
                                                  float*  __restrict__ ct) {
    const int ch = blockIdx.x * 256 + threadIdx.x;   // channel
    const int ij = blockIdx.y;                       // i*G + j
    const int b  = blockIdx.z;
    const long src = ((long)(b * C_DIM + ch)) * (G_DIM * G_DIM) + ij;
    const long dst = ((long)(b * (G_DIM * G_DIM) + ij)) * C_DIM + ch;
    zt[dst] = (__bf16)z[src];
    ct[dst] = c[src];
}

// W (P,C,C) f32 -> bf16, straight copy-convert
__global__ __launch_bounds__(256) void convert_w(const float* __restrict__ W,
                                                 __bf16* __restrict__ wt) {
    const long i = (long)blockIdx.x * 256 + threadIdx.x;
    wt[i] = (__bf16)W[i];
}

// ---------------------------------------------------------------------------
// GEMM: ztwk[k][n, o] = sum_c W[k][o,c] * zt[row(n,k), c]   (bf16 WMMA, f32 acc)
// Block = 128 threads (4 waves). Block tile = 128 rows x 64 cols.
// Wave w computes rows [32w, 32w+32) x 64 cols = 8 x v_wmma_f32_16x16x32_bf16
// per K-step. Double-buffered LDS: global loads for tile i+1 are issued before
// the WMMAs of tile i and committed to the other LDS buffer afterward; one
// barrier per K-step.
// grid = (ceil(maxN/128), C/64, P)
// ---------------------------------------------------------------------------
__global__ __launch_bounds__(128) void gemm_ztwk(const __bf16* __restrict__ zt,
                                                 const __bf16* __restrict__ wt,
                                                 float* __restrict__ ztwk) {
    const int kk = blockIdx.z + 1;                 // prediction step 1..5
    const int Nk = stage_rows(kk);
    const int rowBase = blockIdx.x * 128;
    if (rowBase >= Nk) return;
    const int colBase = blockIdx.y * 64;

    long offs = 0;
    for (int j = 1; j < kk; ++j) offs += (long)stage_rows(j);
    float* outk = ztwk + offs * C_DIM;
    const __bf16* wk = wt + (long)(kk - 1) * C_DIM * C_DIM;

    __shared__ __align__(16) __bf16 aT[2][128 * 32];   // [row][c]  2 x 8 KB
    __shared__ __align__(16) __bf16 bT[2][ 64 * 32];   // [o][c]    2 x 4 KB

    const int t     = threadIdx.x;
    const int lane  = t & 31;
    const int wv    = t >> 5;                      // wave 0..3
    const int rlane = lane & 15;
    const int kg    = lane >> 4;

    // --- loader mapping ---
    // A: thread t loads row t (32 bf16 = 64 B = 4 x int4)
    // B: 2 threads per row, 16 bf16 (32 B = 2 x int4) each
    int n = rowBase + t;
    if (n >= Nk) n = Nk - 1;                       // clamp (stores are guarded)
    const int b  = n & (B_DIM - 1);
    const int jj = (n >> 5) & (G_DIM - 1);
    const int ii = n >> 9;
    const __bf16* arow = zt + (((long)(b * G_DIM + (ii + kk + SKIP)) * G_DIM + jj) * C_DIM);
    const int bhalf = t & 1;
    const __bf16* brow = wk + (long)(colBase + (t >> 1)) * C_DIM + bhalf * 16;

    v8f acc00 = {}, acc01 = {}, acc02 = {}, acc03 = {};
    v8f acc10 = {}, acc11 = {}, acc12 = {}, acc13 = {};

    // --- preload tile 0 into buffer 0 ---
    {
        const int4* sa = (const int4*)(arow);
        int4* da = (int4*)(aT[0] + t * 32);
        da[0] = sa[0]; da[1] = sa[1]; da[2] = sa[2]; da[3] = sa[3];
        const int4* sb = (const int4*)(brow);
        int4* db = (int4*)(bT[0] + (t >> 1) * 32 + bhalf * 16);
        db[0] = sb[0]; db[1] = sb[1];
    }
    __syncthreads();

    const int NITER = C_DIM / 32;                  // 16
    for (int it = 0; it < NITER; ++it) {
        const int cur = it & 1;

        // issue next tile's global loads early (hidden behind WMMAs)
        int4 pa0, pa1, pa2, pa3, pb0, pb1;
        const bool haveNext = (it + 1) < NITER;
        if (haveNext) {
            const int4* sa = (const int4*)(arow + (it + 1) * 32);
            pa0 = sa[0]; pa1 = sa[1]; pa2 = sa[2]; pa3 = sa[3];
            const int4* sb = (const int4*)(brow + (it + 1) * 32);
            pb0 = sb[0]; pb1 = sb[1];
        }

        // --- fragments from current buffer ---
        // A (16x32 bf16, ISA 7.12.2): lane<16: K 0..7 / 16..23;
        // lane>=16: K 8..15 / 24..31 (2 bf16 per VGPR)
        union Frag { unsigned int u[8]; v16bf v; };
        Frag af0, af1;
        const unsigned int* aRowU0 =
            (const unsigned int*)(aT[cur]) + (32 * wv + rlane) * 16;
        const unsigned int* aRowU1 = aRowU0 + 16 * 16;
#pragma unroll
        for (int p = 0; p < 8; ++p) {
            const int idx = (p < 4) ? (p + 4 * kg) : (8 + (p - 4) + 4 * kg);
            af0.u[p] = aRowU0[idx];
            af1.u[p] = aRowU1[idx];
        }

        // B: lane = column o; lanes 0..15 hold K 0..7 / 16..23, lanes 16..31 K 8..15 / 24..31
        Frag bf0, bf1, bf2, bf3;
        const unsigned int* bBase = (const unsigned int*)(bT[cur]);
#pragma unroll
        for (int j = 0; j < 8; ++j) {
            bf0.u[j] = bBase[(0 * 16 + rlane) * 16 + ((j < 4) ? (j + 4 * kg) : (8 + (j - 4) + 4 * kg))];
            bf1.u[j] = bBase[(1 * 16 + rlane) * 16 + ((j < 4) ? (j + 4 * kg) : (8 + (j - 4) + 4 * kg))];
            bf2.u[j] = bBase[(2 * 16 + rlane) * 16 + ((j < 4) ? (j + 4 * kg) : (8 + (j - 4) + 4 * kg))];
            bf3.u[j] = bBase[(3 * 16 + rlane) * 16 + ((j < 4) ? (j + 4 * kg) : (8 + (j - 4) + 4 * kg))];
        }

        acc00 = __builtin_amdgcn_wmma_f32_16x16x32_bf16(false, af0.v, false, bf0.v, (short)0, acc00, false, false);
        acc01 = __builtin_amdgcn_wmma_f32_16x16x32_bf16(false, af0.v, false, bf1.v, (short)0, acc01, false, false);
        acc02 = __builtin_amdgcn_wmma_f32_16x16x32_bf16(false, af0.v, false, bf2.v, (short)0, acc02, false, false);
        acc03 = __builtin_amdgcn_wmma_f32_16x16x32_bf16(false, af0.v, false, bf3.v, (short)0, acc03, false, false);
        acc10 = __builtin_amdgcn_wmma_f32_16x16x32_bf16(false, af1.v, false, bf0.v, (short)0, acc10, false, false);
        acc11 = __builtin_amdgcn_wmma_f32_16x16x32_bf16(false, af1.v, false, bf1.v, (short)0, acc11, false, false);
        acc12 = __builtin_amdgcn_wmma_f32_16x16x32_bf16(false, af1.v, false, bf2.v, (short)0, acc12, false, false);
        acc13 = __builtin_amdgcn_wmma_f32_16x16x32_bf16(false, af1.v, false, bf3.v, (short)0, acc13, false, false);

        // commit next tile into the other buffer
        if (haveNext) {
            int4* da = (int4*)(aT[cur ^ 1] + t * 32);
            da[0] = pa0; da[1] = pa1; da[2] = pa2; da[3] = pa3;
            int4* db = (int4*)(bT[cur ^ 1] + (t >> 1) * 32 + bhalf * 16);
            db[0] = pb0; db[1] = pb1;
        }
        __syncthreads();
    }

    // --- store: VGPR r -> M = r (lanes 0-15) / r+8 (lanes 16-31); N = lane&15 ---
    const int m0   = 8 * kg;
    const int ncol = rlane;
#pragma unroll
    for (int r = 0; r < 8; ++r) {
        const int g0 = rowBase + 32 * wv + m0 + r;
        if (g0 < Nk) {
            float* o = outk + (long)g0 * C_DIM + colBase + ncol;
            o[0]  = acc00[r];
            o[16] = acc01[r];
            o[32] = acc02[r];
            o[48] = acc03[r];
        }
        const int g1 = g0 + 16;
        if (g1 < Nk) {
            float* o = outk + (long)g1 * C_DIM + colBase + ncol;
            o[0]  = acc10[r];
            o[16] = acc11[r];
            o[32] = acc12[r];
            o[48] = acc13[r];
        }
    }
}

// ---------------------------------------------------------------------------
// Scoring: one wave32 per position n. 1 positive + 16 threefry-gathered
// negative dots (len 512), softmax over 17, NLL(target 0), mean-scaled atomic.
// grid = (ceil(maxN/8), 1, P), block = 256 (8 waves)
// ---------------------------------------------------------------------------
__global__ __launch_bounds__(256) void score_nce(const float* __restrict__ ct,
                                                 const float* __restrict__ ztwk,
                                                 float* __restrict__ out) {
    const int kk = blockIdx.z + 1;
    const int Nk = stage_rows(kk);
    long offs = 0;
    for (int j = 1; j < kk; ++j) offs += (long)stage_rows(j);
    const float* zk = ztwk + offs * C_DIM;

    const int wv   = threadIdx.x >> 5;
    const int lane = threadIdx.x & 31;
    const int n    = blockIdx.x * 8 + wv;

    __shared__ float partial[8];
    float myLoss = 0.0f;

    if (n < Nk) {
        const int b  = n & (B_DIM - 1);
        const int jj = (n >> 5) & (G_DIM - 1);
        const int ii = n >> 9;
        const float* ctx = ct + (((long)(b * G_DIM + ii) * G_DIM + jj) * C_DIM);

        float creg[16];
#pragma unroll
        for (int q = 0; q < 16; ++q) creg[q] = ctx[lane + 32 * q];

        float logits[17];
#pragma unroll
        for (int tgt = 0; tgt < 17; ++tgt) {
            int row;
            if (tgt == 0) {
                row = n;
            } else {
                const unsigned int r =
                    threefry2x32(0u, (unsigned int)(1000 + kk),
                                 0u, (unsigned int)(n * NEG + (tgt - 1)));
                row = (int)(r % (unsigned int)Nk);
            }
            const float* zr = zk + (long)row * C_DIM;
            float acc = 0.0f;
#pragma unroll
            for (int q = 0; q < 16; ++q) acc = fmaf(creg[q], zr[lane + 32 * q], acc);
#pragma unroll
            for (int o = 16; o > 0; o >>= 1) acc += __shfl_xor(acc, o, 32);
            logits[tgt] = acc;
        }

        float m = logits[0];
#pragma unroll
        for (int t2 = 1; t2 < 17; ++t2) m = fmaxf(m, logits[t2]);
        float s = 0.0f;
#pragma unroll
        for (int t2 = 0; t2 < 17; ++t2) s += __expf(logits[t2] - m);
        const float p0 = __expf(logits[0] - m) / s;
        myLoss = -__logf(p0 + EPS_F) / ((float)Nk * (float)P_STEPS);
    }

    if (lane == 0) partial[wv] = myLoss;
    __syncthreads();
    if (threadIdx.x == 0) {
        float s = 0.0f;
#pragma unroll
        for (int w = 0; w < 8; ++w) s += partial[w];
        atomicAdd(out, s);
    }
}

// ---------------------------------------------------------------------------
// Launch
// ---------------------------------------------------------------------------
extern "C" void kernel_launch(void* const* d_in, const int* in_sizes, int n_in,
                              void* d_out, int out_size, void* d_ws, size_t ws_size,
                              hipStream_t stream) {
    const float* z = (const float*)d_in[0];
    const float* c = (const float*)d_in[1];
    const float* W = (const float*)d_in[2];
    float* out = (float*)d_out;

    // workspace layout (bytes, 256-aligned)
    char* ws = (char*)d_ws;
    const size_t ZT_BYTES = (size_t)B_DIM * G_DIM * G_DIM * C_DIM * 2;          //  8,388,608
    const size_t WT_BYTES = (size_t)P_STEPS * C_DIM * C_DIM * 2;                //  2,621,440
    const size_t CT_BYTES = (size_t)B_DIM * G_DIM * G_DIM * C_DIM * 4;          // 16,777,216
    __bf16* zt   = (__bf16*)(ws);
    __bf16* wtb  = (__bf16*)(ws + ZT_BYTES);
    float*  ctf  = (float*) (ws + ZT_BYTES + WT_BYTES);
    float*  ztwk = (float*) (ws + ZT_BYTES + WT_BYTES + CT_BYTES);              // 62,914,560

    init_out<<<1, 1, 0, stream>>>(out);

    convert_zc<<<dim3(C_DIM / 256, G_DIM * G_DIM, B_DIM), 256, 0, stream>>>(z, c, zt, ctf);

    const int wElems = P_STEPS * C_DIM * C_DIM;
    convert_w<<<dim3(wElems / 256), 256, 0, stream>>>(W, wtb);

    const int maxN = stage_rows(1);                                   // 7168
    gemm_ztwk<<<dim3((maxN + 127) / 128, C_DIM / 64, P_STEPS), 128, 0, stream>>>(zt, wtb, ztwk);

    score_nce<<<dim3((maxN + 7) / 8, 1, P_STEPS), 256, 0, stream>>>(ctf, ztwk, out);
}
